// MoE_43035572306003
// MI455X (gfx1250) — compile-verified, matrix operand order checked by reading, not run on hardware
//
#include <hip/hip_runtime.h>
#include <math.h>

// Problem dims (fixed by setup_inputs): BS=2, SLEN=2048, DIM=2048, HID=1408, E=8, top_k=2
#define DIM    2048
#define HID    1408
#define NEXP   8
#define TOKENS 4096
#define TOPK   2
#define NSLOT  (TOKENS * TOPK)        // 8192 routed rows
#define BM     128                    // GEMM M tile (expert segments padded to this)
#define CAP    (NSLOT + NEXP * BM)    // 9216 rows incl. padding
#define MT     (CAP / BM)             // 72 M tiles

typedef __bf16 bf16;
typedef __bf16 v16bf __attribute__((ext_vector_type(16)));
typedef __bf16 v8bf  __attribute__((ext_vector_type(8)));
typedef __bf16 v4bf  __attribute__((ext_vector_type(4)));
typedef float  v8f   __attribute__((ext_vector_type(8)));
typedef int    v4i   __attribute__((ext_vector_type(4)));

typedef __attribute__((address_space(1))) v4i* gv4i_p;
typedef __attribute__((address_space(3))) v4i* lv4i_p;

union Frag { v16bf v; v8bf h[2]; };

// ---- CDNA5 async global->LDS copy (ASYNCcnt-tracked DMA), with fallback ----
#if __has_builtin(__builtin_amdgcn_global_load_async_to_lds_b128) && \
    __has_builtin(__builtin_amdgcn_s_wait_asynccnt)
#define HAS_ASYNC_LDS 1
#else
#define HAS_ASYNC_LDS 0
#endif

__device__ __forceinline__ void cp16_to_lds(bf16* ldst, const bf16* gsrc) {
#if HAS_ASYNC_LDS
  __builtin_amdgcn_global_load_async_to_lds_b128((gv4i_p)gsrc, (lv4i_p)ldst, 0, 0);
#else
  *(v8bf*)ldst = *(const v8bf*)gsrc;
#endif
}

__device__ __forceinline__ void async_wait() {
#if HAS_ASYNC_LDS
  __builtin_amdgcn_s_wait_asynccnt(0);
#endif
}

// ---------------------------------------------------------------- init ------
__global__ __launch_bounds__(256) void k_init(float* __restrict__ out,
                                              int* __restrict__ row_token,
                                              int* __restrict__ counts,
                                              int* __restrict__ cursor) {
  int i = blockIdx.x * 256 + threadIdx.x;
  if (i < TOKENS * DIM) out[i] = 0.0f;
  if (i < CAP) row_token[i] = -1;
  if (i < NEXP) { counts[i] = 0; cursor[i] = 0; }
}

// -------------------------------------------------------------- router ------
__global__ __launch_bounds__(256) void k_router(const float* __restrict__ x,
                                                const float* __restrict__ gw,
                                                float* __restrict__ top_scores,
                                                int* __restrict__ sel,
                                                int* __restrict__ counts) {
  int wave = threadIdx.x >> 5;
  int lane = threadIdx.x & 31;
  int t = blockIdx.x * 8 + wave;
  if (t >= TOKENS) return;

  float acc[NEXP];
#pragma unroll
  for (int e = 0; e < NEXP; e++) acc[e] = 0.0f;

  const float* xr = x + (size_t)t * DIM;
  for (int d = lane; d < DIM; d += 32) {
    float xv = xr[d];
#pragma unroll
    for (int e = 0; e < NEXP; e++) acc[e] += xv * gw[e * DIM + d];
  }
#pragma unroll
  for (int e = 0; e < NEXP; e++) {
#pragma unroll
    for (int off = 16; off >= 1; off >>= 1) acc[e] += __shfl_xor(acc[e], off, 32);
  }
  if (lane == 0) {
    float s[NEXP];
#pragma unroll
    for (int e = 0; e < NEXP; e++) s[e] = 1.0f / (1.0f + __expf(-acc[e]));
    int b0 = 0; float v0 = s[0];
    for (int e = 1; e < NEXP; e++) if (s[e] > v0) { v0 = s[e]; b0 = e; }   // lowest idx wins ties
    int b1 = -1; float v1 = -1e30f;
    for (int e = 0; e < NEXP; e++) if (e != b0 && s[e] > v1) { v1 = s[e]; b1 = e; }
    sel[t * 2 + 0] = b0; sel[t * 2 + 1] = b1;
    top_scores[t * 2 + 0] = v0; top_scores[t * 2 + 1] = v1;
    atomicAdd(&counts[b0], 1);
    atomicAdd(&counts[b1], 1);
  }
}

// --------------------------------------------------------------- setup ------
__global__ void k_setup(const int* __restrict__ counts,
                        int* __restrict__ pbase,
                        int* __restrict__ tile_expert) {
  if (threadIdx.x == 0 && blockIdx.x == 0) {
    int b = 0;
    for (int e = 0; e < NEXP; e++) {
      pbase[e] = b;
      b += ((counts[e] + BM - 1) / BM) * BM;
    }
    pbase[NEXP] = b;
    for (int tl = 0; tl < MT; tl++) {
      int r = tl * BM;
      int e = -1;
      for (int i = 0; i < NEXP; i++)
        if (r >= pbase[i] && r < pbase[i + 1]) e = i;
      tile_expert[tl] = e;
    }
  }
}

// --------------------------------------------------------------- place ------
__global__ __launch_bounds__(256) void k_place(const int* __restrict__ sel,
                                               const float* __restrict__ top_scores,
                                               const int* __restrict__ pbase,
                                               int* __restrict__ cursor,
                                               int* __restrict__ row_token,
                                               float* __restrict__ row_score) {
  int slot = blockIdx.x * 256 + threadIdx.x;
  if (slot >= NSLOT) return;
  int e = sel[slot];
  int p = pbase[e] + atomicAdd(&cursor[e], 1);
  row_token[p] = slot;
  row_score[p] = top_scores[slot];
}

// -------------------------------------------------------------- gather ------
__global__ __launch_bounds__(256) void k_gather(const float* __restrict__ x,
                                                const int* __restrict__ row_token,
                                                const float* __restrict__ row_score,
                                                bf16* __restrict__ Xr) {
  int pos = blockIdx.x;
  int ts = row_token[pos];
  float sc = (ts >= 0) ? row_score[pos] : 0.0f;
  const float* src = x + (size_t)((ts >= 0) ? (ts >> 1) : 0) * DIM;
  bf16* dst = Xr + (size_t)pos * DIM;
  for (int d = threadIdx.x; d < DIM; d += 256) {
    float v = (ts >= 0) ? src[d] * sc : 0.0f;
    dst[d] = (bf16)v;
  }
}

// --------------------------------------------------------------- GEMM1 ------
// H = silu(Xr @ W1^T) * (Xr @ W3^T). 128x64 block tile, 8 waves, double-buffered LDS.
__global__ __launch_bounds__(256) void k_gemm1(const bf16* __restrict__ Xr,
                                               const float* __restrict__ W1,
                                               const float* __restrict__ W3,
                                               const int* __restrict__ tile_expert,
                                               bf16* __restrict__ H) {
  int nb  = blockIdx.x;            // 0..HID/64-1
  int mtl = blockIdx.y;            // 0..MT-1
  int e = tile_expert[mtl];
  if (e < 0) return;

  __shared__ bf16 sA [2][BM][32];
  __shared__ bf16 sB1[2][64][32];
  __shared__ bf16 sB3[2][64][32];

  int tid = threadIdx.x;
  int wave = tid >> 5, lane = tid & 31;
  int wm = wave & 3, wn = wave >> 2;       // wave covers 32x32 of the 128x64 tile
  int hlf = lane >> 4, l16 = lane & 15;

  v8f acc1[2][2] = {};
  v8f acc3[2][2] = {};

  const bf16*  xa  = Xr + (size_t)mtl * BM * DIM;
  const float* w1e = W1 + ((size_t)e * HID + (size_t)nb * 64) * DIM;
  const float* w3e = W3 + ((size_t)e * HID + (size_t)nb * 64) * DIM;

  // Straight-line staging (no masked loops): each thread owns 2 A-rows / 2 B-rows.
  const int ra = tid >> 2, ca = (tid & 3) * 8;   // A: 128 rows x 32 k, 16B chunks
  const int rb = tid >> 3, cb = (tid & 7) * 4;   // B: 64 rows x 32 k, 4 f32 chunks

  auto stage = [&](int buf, int k0) {
    cp16_to_lds(&sA[buf][ra][ca],      xa + (size_t)ra * DIM + k0 + ca);
    cp16_to_lds(&sA[buf][ra + 64][ca], xa + (size_t)(ra + 64) * DIM + k0 + ca);
    float4 a0 = *(const float4*)(w1e + (size_t)rb * DIM + k0 + cb);
    float4 a1 = *(const float4*)(w1e + (size_t)(rb + 32) * DIM + k0 + cb);
    float4 b0 = *(const float4*)(w3e + (size_t)rb * DIM + k0 + cb);
    float4 b1 = *(const float4*)(w3e + (size_t)(rb + 32) * DIM + k0 + cb);
    *(v4bf*)&sB1[buf][rb][cb]      = (v4bf){(bf16)a0.x, (bf16)a0.y, (bf16)a0.z, (bf16)a0.w};
    *(v4bf*)&sB1[buf][rb + 32][cb] = (v4bf){(bf16)a1.x, (bf16)a1.y, (bf16)a1.z, (bf16)a1.w};
    *(v4bf*)&sB3[buf][rb][cb]      = (v4bf){(bf16)b0.x, (bf16)b0.y, (bf16)b0.z, (bf16)b0.w};
    *(v4bf*)&sB3[buf][rb + 32][cb] = (v4bf){(bf16)b1.x, (bf16)b1.y, (bf16)b1.z, (bf16)b1.w};
  };

  auto compute = [&](int buf) {
    // 16-bit A striping (lane m: K 0..7,16..23 ; lane m+16: K 8..15,24..31)
    Frag af[2], b1f[2], b3f[2];
#pragma unroll
    for (int i = 0; i < 2; i++) {
      int r = wm * 32 + i * 16 + l16;
      af[i].h[0] = *(const v8bf*)&sA[buf][r][hlf * 8];
      af[i].h[1] = *(const v8bf*)&sA[buf][r][16 + hlf * 8];
    }
#pragma unroll
    for (int j = 0; j < 2; j++) {
      int cix = wn * 32 + j * 16 + l16;
      b1f[j].h[0] = *(const v8bf*)&sB1[buf][cix][hlf * 8];
      b1f[j].h[1] = *(const v8bf*)&sB1[buf][cix][16 + hlf * 8];
      b3f[j].h[0] = *(const v8bf*)&sB3[buf][cix][hlf * 8];
      b3f[j].h[1] = *(const v8bf*)&sB3[buf][cix][16 + hlf * 8];
    }
#pragma unroll
    for (int i = 0; i < 2; i++)
#pragma unroll
      for (int j = 0; j < 2; j++) {
        acc1[i][j] = __builtin_amdgcn_wmma_f32_16x16x32_bf16(
            false, af[i].v, false, b1f[j].v, (short)0, acc1[i][j], false, false);
        acc3[i][j] = __builtin_amdgcn_wmma_f32_16x16x32_bf16(
            false, af[i].v, false, b3f[j].v, (short)0, acc3[i][j], false, false);
      }
  };

  stage(0, 0);
  async_wait();
  __syncthreads();
  int cur = 0;
  for (int k0 = 0; k0 < DIM; k0 += 32) {
    int nxt = cur ^ 1;
    if (k0 + 32 < DIM) stage(nxt, k0 + 32);
    compute(cur);
    async_wait();
    __syncthreads();
    cur = nxt;
  }

  // SwiGLU epilogue -> bf16 H. D layout: lane l16 = N, vgpr r -> M = r + 8*hlf
#pragma unroll
  for (int i = 0; i < 2; i++)
#pragma unroll
    for (int j = 0; j < 2; j++)
#pragma unroll
      for (int r = 0; r < 8; r++) {
        float g = acc1[i][j][r];
        float u = acc3[i][j][r];
        float hv = (g / (1.0f + __expf(-g))) * u;     // silu(g) * u
        int row = mtl * BM + wm * 32 + i * 16 + r + 8 * hlf;
        int col = nb * 64 + wn * 32 + j * 16 + l16;
        H[(size_t)row * HID + col] = (bf16)hv;
      }
}

// --------------------------------------------------------------- GEMM2 ------
// out[token] += H @ W2^T  (atomic f32 scatter combines the two top-k slots)
__global__ __launch_bounds__(256) void k_gemm2(const bf16* __restrict__ H,
                                               const float* __restrict__ W2,
                                               const int* __restrict__ tile_expert,
                                               const int* __restrict__ row_token,
                                               float* __restrict__ out) {
  int nb  = blockIdx.x;            // 0..DIM/64-1
  int mtl = blockIdx.y;
  int e = tile_expert[mtl];
  if (e < 0) return;

  __shared__ bf16 sA[2][BM][32];
  __shared__ bf16 sB[2][64][32];

  int tid = threadIdx.x;
  int wave = tid >> 5, lane = tid & 31;
  int wm = wave & 3, wn = wave >> 2;
  int hlf = lane >> 4, l16 = lane & 15;

  v8f acc[2][2] = {};

  const bf16*  ha  = H  + (size_t)mtl * BM * HID;
  const float* w2e = W2 + ((size_t)e * DIM + (size_t)nb * 64) * HID;

  const int ra = tid >> 2, ca = (tid & 3) * 8;
  const int rb = tid >> 3, cb = (tid & 7) * 4;

  auto stage = [&](int buf, int k0) {
    cp16_to_lds(&sA[buf][ra][ca],      ha + (size_t)ra * HID + k0 + ca);
    cp16_to_lds(&sA[buf][ra + 64][ca], ha + (size_t)(ra + 64) * HID + k0 + ca);
    float4 a0 = *(const float4*)(w2e + (size_t)rb * HID + k0 + cb);
    float4 a1 = *(const float4*)(w2e + (size_t)(rb + 32) * HID + k0 + cb);
    *(v4bf*)&sB[buf][rb][cb]      = (v4bf){(bf16)a0.x, (bf16)a0.y, (bf16)a0.z, (bf16)a0.w};
    *(v4bf*)&sB[buf][rb + 32][cb] = (v4bf){(bf16)a1.x, (bf16)a1.y, (bf16)a1.z, (bf16)a1.w};
  };

  auto compute = [&](int buf) {
    Frag af[2], bfr[2];
#pragma unroll
    for (int i = 0; i < 2; i++) {
      int r = wm * 32 + i * 16 + l16;
      af[i].h[0] = *(const v8bf*)&sA[buf][r][hlf * 8];
      af[i].h[1] = *(const v8bf*)&sA[buf][r][16 + hlf * 8];
    }
#pragma unroll
    for (int j = 0; j < 2; j++) {
      int cix = wn * 32 + j * 16 + l16;
      bfr[j].h[0] = *(const v8bf*)&sB[buf][cix][hlf * 8];
      bfr[j].h[1] = *(const v8bf*)&sB[buf][cix][16 + hlf * 8];
    }
#pragma unroll
    for (int i = 0; i < 2; i++)
#pragma unroll
      for (int j = 0; j < 2; j++)
        acc[i][j] = __builtin_amdgcn_wmma_f32_16x16x32_bf16(
            false, af[i].v, false, bfr[j].v, (short)0, acc[i][j], false, false);
  };

  stage(0, 0);
  async_wait();
  __syncthreads();
  int cur = 0;
  for (int k0 = 0; k0 < HID; k0 += 32) {
    int nxt = cur ^ 1;
    if (k0 + 32 < HID) stage(nxt, k0 + 32);
    compute(cur);
    async_wait();
    __syncthreads();
    cur = nxt;
  }

#pragma unroll
  for (int i = 0; i < 2; i++)
#pragma unroll
    for (int r = 0; r < 8; r++) {
      int row = mtl * BM + wm * 32 + i * 16 + r + 8 * hlf;
      int ts = row_token[row];
      if (ts < 0) continue;                       // padding row
      int t = ts >> 1;
#pragma unroll
      for (int j = 0; j < 2; j++) {
        int col = nb * 64 + wn * 32 + j * 16 + l16;
        atomicAdd(&out[(size_t)t * DIM + col], acc[i][j][r]);
      }
    }
}

// -------------------------------------------------------------- launch ------
extern "C" void kernel_launch(void* const* d_in, const int* in_sizes, int n_in,
                              void* d_out, int out_size, void* d_ws, size_t ws_size,
                              hipStream_t stream) {
  const float* x  = (const float*)d_in[0];
  const float* gw = (const float*)d_in[1];
  const float* w1 = (const float*)d_in[2];
  const float* w2 = (const float*)d_in[3];
  const float* w3 = (const float*)d_in[4];
  float* out = (float*)d_out;

  char* ws = (char*)d_ws;
  size_t off = 0;
  auto alloc = [&](size_t bytes) -> char* {
    char* p = ws + off;
    off = (off + bytes + 255) & ~(size_t)255;
    return p;
  };
  bf16*  Xr         = (bf16*) alloc((size_t)CAP * DIM * sizeof(bf16));   // 37.7 MB
  bf16*  Hb         = (bf16*) alloc((size_t)CAP * HID * sizeof(bf16));   // 26.0 MB
  float* top_scores = (float*)alloc((size_t)NSLOT * sizeof(float));
  int*   sel        = (int*)  alloc((size_t)NSLOT * sizeof(int));
  int*   row_token  = (int*)  alloc((size_t)CAP * sizeof(int));
  float* row_score  = (float*)alloc((size_t)CAP * sizeof(float));
  int*   counts     = (int*)  alloc(NEXP * sizeof(int));
  int*   cursor     = (int*)  alloc(NEXP * sizeof(int));
  int*   pbase      = (int*)  alloc((NEXP + 1) * sizeof(int));
  int*   tile_exp   = (int*)  alloc(MT * sizeof(int));

  k_init  <<<(TOKENS * DIM + 255) / 256, 256, 0, stream>>>(out, row_token, counts, cursor);
  k_router<<<TOKENS / 8,               256, 0, stream>>>(x, gw, top_scores, sel, counts);
  k_setup <<<1,                          1, 0, stream>>>(counts, pbase, tile_exp);
  k_place <<<NSLOT / 256,              256, 0, stream>>>(sel, top_scores, pbase, cursor, row_token, row_score);
  k_gather<<<CAP,                      256, 0, stream>>>(x, row_token, row_score, Xr);
  k_gemm1 <<<dim3(HID / 64, MT),       256, 0, stream>>>(Xr, w1, w3, tile_exp, Hb);
  k_gemm2 <<<dim3(DIM / 64, MT),       256, 0, stream>>>(Hb, w2, tile_exp, row_token, out);

  (void)in_sizes; (void)n_in; (void)out_size; (void)ws_size;
}
